// SingleStandRwkvBlock_6803228196959
// MI455X (gfx1250) — compile-verified, matrix operand order checked by reading, not run on hardware
//
#include <hip/hip_runtime.h>
#include <hip/hip_bf16.h>

#define DEV __device__ __forceinline__

// ---------------------------------------------------------------------------
// Types for CDNA5 WMMA (wave32): bf16 A/B fragments (16 x bf16 = 8 VGPRs),
// f32 accumulator (8 VGPRs).
// ---------------------------------------------------------------------------
typedef __attribute__((ext_vector_type(16))) __bf16 bf16x16;
typedef __attribute__((ext_vector_type(8)))  float  f32x8;

union Frag { bf16x16 v; uint4 q[2]; };

DEV unsigned short f32_to_bf16(float f) {
  union { float f; unsigned int u; } x; x.f = f;
  unsigned int u = x.u;
  return (unsigned short)((u + 0x7FFFu + ((u >> 16) & 1u)) >> 16);  // RNE
}

// Flat LDS pointers carry the wave-relative LDS offset in bits [31:0]
// (ISA 10.2: LDS aperture decode truncates to addr[31:0]).
DEV unsigned int lds_off(const void* p) {
  return (unsigned int)(uintptr_t)p;
}

// CDNA5 async direct-to-LDS copy (VGLOBAL opcode 98, tracked by ASYNCcnt):
// 16 bytes per lane, memory -> LDS without a VGPR round trip.
DEV void async_load_b128(unsigned int dst_lds, const void* src) {
  asm volatile("global_load_async_to_lds_b128 %0, %1, off"
               :
               : "v"(dst_lds), "v"((unsigned long long)(uintptr_t)src)
               : "memory");
}

DEV void wait_async0() {
  asm volatile("s_wait_asynccnt 0" ::: "memory");
}

// ---------------------------------------------------------------------------
// Weight convert + transpose: in[R][C] f32  ->  out[C][R] bf16
// 32x32 tiles through padded LDS; blockDim = (32, 8).
// ---------------------------------------------------------------------------
__global__ __launch_bounds__(256)
void convert_transpose(const float* __restrict__ in, unsigned short* __restrict__ out,
                       int R, int C) {
  __shared__ unsigned short tile[32][33];
  const int bx = blockIdx.x, by = blockIdx.y;
  const int tx = threadIdx.x, ty = threadIdx.y;
#pragma unroll
  for (int j = 0; j < 4; ++j) {
    int r = by * 32 + ty + j * 8, c = bx * 32 + tx;
    tile[ty + j * 8][tx] = f32_to_bf16(in[(size_t)r * C + c]);
  }
  __syncthreads();
#pragma unroll
  for (int j = 0; j < 4; ++j) {
    int c = bx * 32 + ty + j * 8, r = by * 32 + tx;
    out[(size_t)c * R + r] = tile[tx][ty + j * 8];
  }
}

// ---------------------------------------------------------------------------
// Block reduction helpers (wave32 shuffles + cross-wave via LDS), 256 threads.
// ---------------------------------------------------------------------------
DEV float block_sum(float v, float* sred) {
#pragma unroll
  for (int off = 16; off > 0; off >>= 1) v += __shfl_xor(v, off, 32);
  const int tid = threadIdx.x;
  __syncthreads();                 // protect sred from previous use
  if ((tid & 31) == 0) sred[tid >> 5] = v;
  __syncthreads();
  float t = 0.f;
#pragma unroll
  for (int i = 0; i < 8; ++i) t += sred[i];
  return t;
}

// LayerNorm of one H=2048 row held as 8 regs/thread (256 threads).
DEV void row_ln(float (&v)[8], const float* __restrict__ w, const float* __restrict__ b,
                float* sred, int tid, float invH) {
  float s = 0.f;
#pragma unroll
  for (int i = 0; i < 8; ++i) s += v[i];
  const float mu = block_sum(s, sred) * invH;
  float q = 0.f;
#pragma unroll
  for (int i = 0; i < 8; ++i) { float d = v[i] - mu; q += d * d; }
  const float var = block_sum(q, sred) * invH;
  const float inv = rsqrtf(var + 1e-5f);
#pragma unroll
  for (int i = 0; i < 8; ++i) {
    int idx = i * 256 + tid;
    v[i] = (v[i] - mu) * inv * w[idx] + b[idx];
  }
}

// ---------------------------------------------------------------------------
// LN (+optional second LN) -> token shift -> mix -> bf16 outputs.
// One block per row (b,t). H must be 2048 (8 vals/thread @ 256 threads).
// ---------------------------------------------------------------------------
template <bool TWO_LN>
__global__ __launch_bounds__(256)
void ln_shift_mix(const float* __restrict__ xin,
                  const float* __restrict__ w1, const float* __restrict__ b1,
                  const float* __restrict__ w2, const float* __restrict__ b2,
                  const float* __restrict__ mk, const float* __restrict__ mv,
                  const float* __restrict__ mr,
                  float* __restrict__ ln1_out,
                  unsigned short* __restrict__ okx,
                  unsigned short* __restrict__ ovx,
                  unsigned short* __restrict__ orx,
                  int T, int H) {
  __shared__ float sred[8];
  const int row = blockIdx.x;
  const int t = row % T;
  const int tid = threadIdx.x;
  const float invH = 1.0f / (float)H;

  float cur[8];
#pragma unroll
  for (int i = 0; i < 8; ++i) cur[i] = xin[(size_t)row * H + i * 256 + tid];
  row_ln(cur, w1, b1, sred, tid, invH);
  if (TWO_LN) {
#pragma unroll
    for (int i = 0; i < 8; ++i) ln1_out[(size_t)row * H + i * 256 + tid] = cur[i];
    row_ln(cur, w2, b2, sred, tid, invH);
  }

  float prev[8];
  if (t > 0) {
#pragma unroll
    for (int i = 0; i < 8; ++i) prev[i] = xin[(size_t)(row - 1) * H + i * 256 + tid];
    row_ln(prev, w1, b1, sred, tid, invH);
    if (TWO_LN) row_ln(prev, w2, b2, sred, tid, invH);
  } else {
#pragma unroll
    for (int i = 0; i < 8; ++i) prev[i] = 0.f;
  }

#pragma unroll
  for (int i = 0; i < 8; ++i) {
    const int idx = i * 256 + tid;
    const size_t o = (size_t)row * H + idx;
    float a = mk[idx];
    okx[o] = f32_to_bf16(cur[i] * a + prev[i] * (1.f - a));
    if (ovx) {
      float c = mv[idx];
      ovx[o] = f32_to_bf16(cur[i] * c + prev[i] * (1.f - c));
    }
    float r = mr[idx];
    orx[o] = f32_to_bf16(cur[i] * r + prev[i] * (1.f - r));
  }
}

// ---------------------------------------------------------------------------
// bf16 WMMA GEMM: C[M][N] = A[M][K] x Bt[N][K]^T  (Bt is B transposed).
// 128x128 block tile, 8 waves (4x2), 32x64 per wave = 8 WMMAs per 32-deep
// K stage. Double-buffered LDS filled by GLOBAL_LOAD_ASYNC_TO_LDS_B128 so
// the copy of tile t+1 overlaps the WMMAs on tile t (ASYNCcnt, no VGPR hop).
// LDS rows padded to 80B so fragment ds_load_b128 stay 16B-aligned.
// ---------------------------------------------------------------------------
enum { EPI_STORE = 0, EPI_SIGMOID = 1, EPI_RELUSQ_BF16 = 2, EPI_ADD = 3, EPI_ADD_MUL = 4 };

template <int EPI>
__global__ __launch_bounds__(256)
void gemm_bf16_wmma(const unsigned short* __restrict__ A,
                    const unsigned short* __restrict__ Bt,
                    float* __restrict__ Cf, unsigned short* __restrict__ Cb,
                    const float* __restrict__ base, const float* __restrict__ mul,
                    int M, int N, int K) {
  constexpr int LDT = 40;  // shorts per LDS row (80 bytes, 16B-aligned rows)
  __shared__ __align__(16) unsigned short As[2][128 * LDT];
  __shared__ __align__(16) unsigned short Bs[2][128 * LDT];   // 40KB total

  const int tid = threadIdx.x;
  const int wave = tid >> 5, lane = tid & 31;
  const int wr = wave >> 1;          // 0..3 -> 32-row group
  const int wc = wave & 1;           // 0..1 -> 64-col group
  const int m0 = blockIdx.y * 128;
  const int n0 = blockIdx.x * 128;

  // Staging: 512 16B-chunks per 128x32 tile; each thread owns 2 chunks of A
  // and 2 of B (rows r0 and r0+64 at the same 8-element column).
  const int r0 = tid >> 2;
  const int c0 = (tid & 3) * 8;
  const unsigned short* gA = A + (size_t)(m0 + r0) * K + c0;
  const unsigned short* gB = Bt + (size_t)(n0 + r0) * K + c0;
  const size_t rowskip = (size_t)64 * K;

  f32x8 acc[2][4] = {};

  const int nk = K / 32;
  // prologue: tile 0 -> buffer 0
  {
    const unsigned int la = lds_off(&As[0][r0 * LDT + c0]);
    const unsigned int lb = lds_off(&Bs[0][r0 * LDT + c0]);
    async_load_b128(la, gA);
    async_load_b128(la + 64 * LDT * 2, gA + rowskip);
    async_load_b128(lb, gB);
    async_load_b128(lb + 64 * LDT * 2, gB + rowskip);
  }

  for (int t = 0; t < nk; ++t) {
    wait_async0();       // this wave's chunks of tile t are in LDS
    __syncthreads();     // everyone's chunks are in LDS

    if (t + 1 < nk) {    // kick off tile t+1 into the other buffer
      const int kb = (t + 1) * 32;
      const int bi = (t + 1) & 1;
      const unsigned int la = lds_off(&As[bi][r0 * LDT + c0]);
      const unsigned int lb = lds_off(&Bs[bi][r0 * LDT + c0]);
      async_load_b128(la, gA + kb);
      async_load_b128(la + 64 * LDT * 2, gA + rowskip + kb);
      async_load_b128(lb, gB + kb);
      async_load_b128(lb + 64 * LDT * 2, gB + rowskip + kb);
      if (t + 2 < nk) {  // warm L2 for tile t+2 (global_prefetch_b8)
        __builtin_prefetch(gA + kb + 32, 0, 1);
        __builtin_prefetch(gB + kb + 32, 0, 1);
      }
    }

    // Per-lane fragment layout (16-bit A-matrix 16x32): lanes 0-15 hold
    // K in [0,8)u[16,24), lanes 16-31 hold K in [8,16)u[24,32) of row lane&15.
    const unsigned short* as = As[t & 1];
    const unsigned short* bs = Bs[t & 1];
    const int frow = lane & 15;
    const int kh = (lane >> 4) * 8;
    Frag a[2], b[4];
#pragma unroll
    for (int mi = 0; mi < 2; ++mi) {
      const unsigned short* p = as + (wr * 32 + mi * 16 + frow) * LDT + kh;
      a[mi].q[0] = *(const uint4*)p;
      a[mi].q[1] = *(const uint4*)(p + 16);
    }
#pragma unroll
    for (int ni = 0; ni < 4; ++ni) {
      const unsigned short* p = bs + (wc * 64 + ni * 16 + frow) * LDT + kh;
      b[ni].q[0] = *(const uint4*)p;
      b[ni].q[1] = *(const uint4*)(p + 16);
    }
#pragma unroll
    for (int mi = 0; mi < 2; ++mi)
#pragma unroll
      for (int ni = 0; ni < 4; ++ni)
        acc[mi][ni] = __builtin_amdgcn_wmma_f32_16x16x32_bf16(
            false, a[mi].v, false, b[ni].v, (short)0, acc[mi][ni], false, false);

    __syncthreads();     // all reads of buf t done before iter t+1 loads tile t+2 into it
  }

  // Epilogue. C/D layout: acc[r] @ lane -> m = r + 8*(lane>=16), n = lane&15.
  const int mo = (lane >> 4) * 8;
  const int nn = lane & 15;
#pragma unroll
  for (int mi = 0; mi < 2; ++mi)
#pragma unroll
    for (int ni = 0; ni < 4; ++ni) {
      const int crow0 = m0 + wr * 32 + mi * 16 + mo;
      const int ccol = n0 + wc * 64 + ni * 16 + nn;
#pragma unroll
      for (int r = 0; r < 8; ++r) {
        const size_t idx = (size_t)(crow0 + r) * N + ccol;
        float v = acc[mi][ni][r];
        if (EPI == EPI_STORE) {
          Cf[idx] = v;
        } else if (EPI == EPI_SIGMOID) {
          Cf[idx] = 1.0f / (1.0f + __expf(-v));
        } else if (EPI == EPI_RELUSQ_BF16) {
          float t = fmaxf(v, 0.f);
          Cb[idx] = f32_to_bf16(t * t);
        } else if (EPI == EPI_ADD) {
          Cf[idx] = base[idx] + v;
        } else {  // EPI_ADD_MUL
          Cf[idx] = base[idx] + mul[idx] * v;
        }
      }
    }
}

// ---------------------------------------------------------------------------
// WKV recurrence: one lane per (b, a) channel, sequential over T.
// ---------------------------------------------------------------------------
__global__ __launch_bounds__(256)
void wkv_scan(const float* __restrict__ k, const float* __restrict__ v,
              const float* __restrict__ time_decay, const float* __restrict__ time_first,
              float* __restrict__ wkv, int B, int T, int A) {
  const int g = blockIdx.x * 256 + threadIdx.x;
  if (g >= B * A) return;
  const int b = g / A, a = g % A;
  const float w = -__expf(time_decay[a]);
  const float tf = time_first[a];
  float aa = 0.f, bb = 0.f, pp = -1e38f;
  const size_t rowbase = (size_t)b * T * A + a;
  for (int t = 0; t < T; ++t) {
    const size_t o = rowbase + (size_t)t * A;
    const float kk = k[o], vv = v[o];
    float ww = tf + kk;
    float q = fmaxf(pp, ww);
    float e1 = __expf(pp - q), e2 = __expf(ww - q);
    wkv[o] = (e1 * aa + e2 * vv) / (e1 * bb + e2);
    ww = w + pp;
    q = fmaxf(ww, kk);
    e1 = __expf(ww - q);
    e2 = __expf(kk - q);
    aa = e1 * aa + e2 * vv;
    bb = e1 * bb + e2;
    pp = q;
  }
}

__global__ __launch_bounds__(256)
void mul_to_bf16(const float* __restrict__ a, const float* __restrict__ b,
                 unsigned short* __restrict__ out, size_t n) {
  const size_t i = (size_t)blockIdx.x * 256 + threadIdx.x;
  if (i < n) out[i] = f32_to_bf16(a[i] * b[i]);
}

// ---------------------------------------------------------------------------
// Host orchestration
// ---------------------------------------------------------------------------
extern "C" void kernel_launch(void* const* d_in, const int* in_sizes, int n_in,
                              void* d_out, int out_size, void* d_ws, size_t ws_size,
                              hipStream_t stream) {
  constexpr int B = 2, T = 2048, H = 2048, A = 2048, F = 8192;
  constexpr int M = B * T;

  const float* hidden     = (const float*)d_in[0];
  const float* pre_w      = (const float*)d_in[1];
  const float* pre_b      = (const float*)d_in[2];
  const float* ln1_w      = (const float*)d_in[3];
  const float* ln1_b      = (const float*)d_in[4];
  const float* ln2_w      = (const float*)d_in[5];
  const float* ln2_b      = (const float*)d_in[6];
  const float* time_decay = (const float*)d_in[7];
  const float* time_first = (const float*)d_in[8];
  const float* tmk        = (const float*)d_in[9];
  const float* tmv        = (const float*)d_in[10];
  const float* tmr        = (const float*)d_in[11];
  const float* Wk         = (const float*)d_in[12];
  const float* Wv         = (const float*)d_in[13];
  const float* Wr         = (const float*)d_in[14];
  const float* Wo         = (const float*)d_in[15];
  const float* f_tmk      = (const float*)d_in[16];
  const float* f_tmr      = (const float*)d_in[17];
  const float* f_Wk       = (const float*)d_in[18];
  const float* f_Wr       = (const float*)d_in[19];
  const float* f_Wv       = (const float*)d_in[20];

  // --- workspace carve-up (all sizes are multiples of 256B) ---
  char* w = (char*)d_ws;
  size_t off = 0;
  auto take = [&](size_t bytes) -> void* {
    void* p = w + off;
    off += (bytes + 255) & ~(size_t)255;
    return p;
  };
  unsigned short* WkT  = (unsigned short*)take((size_t)A * H * 2);  // [A][H]
  unsigned short* WvT  = (unsigned short*)take((size_t)A * H * 2);
  unsigned short* WrT  = (unsigned short*)take((size_t)A * H * 2);
  unsigned short* WoT  = (unsigned short*)take((size_t)H * A * 2);  // [H][A]
  unsigned short* fWkT = (unsigned short*)take((size_t)F * H * 2);  // [F][H]
  unsigned short* fWrT = (unsigned short*)take((size_t)H * H * 2);  // [H][H]
  unsigned short* fWvT = (unsigned short*)take((size_t)H * F * 2);  // [H][F]
  float*          h0   = (float*)take((size_t)M * H * 4);
  unsigned short* kx   = (unsigned short*)take((size_t)M * H * 2);
  unsigned short* vx   = (unsigned short*)take((size_t)M * H * 2);
  unsigned short* rx   = (unsigned short*)take((size_t)M * H * 2);
  float*          kbuf = (float*)take((size_t)M * A * 4);
  float*          vbuf = (float*)take((size_t)M * A * 4);
  float*          rbuf = (float*)take((size_t)M * A * 4);
  unsigned short* kk2  = (unsigned short*)take((size_t)M * F * 2);
  // lifetime-safe aliases (regions dead at reuse time):
  float*          wkv  = (float*)kx;                                       // 32MB over kx+vx
  unsigned short* rwkv = rx;                                               // 16MB
  float*          h1   = kbuf;                                             // 32MB
  unsigned short* kx2  = (unsigned short*)vbuf;                            // 16MB
  unsigned short* rx2  = (unsigned short*)((char*)vbuf + (size_t)M * H * 2);
  float*          r2   = rbuf;                                             // 32MB
  float*          out  = (float*)d_out;

  const dim3 tb(32, 8);
  // 1) weights -> bf16, transposed to [N][K]
  convert_transpose<<<dim3(A / 32, H / 32), tb, 0, stream>>>(Wk, WkT, H, A);
  convert_transpose<<<dim3(A / 32, H / 32), tb, 0, stream>>>(Wv, WvT, H, A);
  convert_transpose<<<dim3(A / 32, H / 32), tb, 0, stream>>>(Wr, WrT, H, A);
  convert_transpose<<<dim3(H / 32, A / 32), tb, 0, stream>>>(Wo, WoT, A, H);
  convert_transpose<<<dim3(F / 32, H / 32), tb, 0, stream>>>(f_Wk, fWkT, H, F);
  convert_transpose<<<dim3(H / 32, H / 32), tb, 0, stream>>>(f_Wr, fWrT, H, H);
  convert_transpose<<<dim3(H / 32, F / 32), tb, 0, stream>>>(f_Wv, fWvT, F, H);

  // 2) pre_ln -> ln1 -> shift -> mixes (kx/vx/rx bf16), store h0
  ln_shift_mix<true><<<M, 256, 0, stream>>>(hidden, pre_w, pre_b, ln1_w, ln1_b,
                                            tmk, tmv, tmr, h0, kx, vx, rx, T, H);

  // 3-5) r = sigmoid(rx@Wr), k = kx@Wk, v = vx@Wv
  gemm_bf16_wmma<EPI_SIGMOID><<<dim3(A / 128, M / 128), 256, 0, stream>>>(
      rx, WrT, rbuf, nullptr, nullptr, nullptr, M, A, H);
  gemm_bf16_wmma<EPI_STORE><<<dim3(A / 128, M / 128), 256, 0, stream>>>(
      kx, WkT, kbuf, nullptr, nullptr, nullptr, M, A, H);
  gemm_bf16_wmma<EPI_STORE><<<dim3(A / 128, M / 128), 256, 0, stream>>>(
      vx, WvT, vbuf, nullptr, nullptr, nullptr, M, A, H);

  // 6) sequential WKV recurrence (one lane per channel)
  wkv_scan<<<(B * A + 255) / 256, 256, 0, stream>>>(kbuf, vbuf, time_decay,
                                                    time_first, wkv, B, T, A);

  // 7) rwkv = bf16(r * wkv)
  mul_to_bf16<<<(int)(((size_t)M * A + 255) / 256), 256, 0, stream>>>(
      rbuf, wkv, rwkv, (size_t)M * A);

  // 8) h1 = h0 + rwkv @ Wo
  gemm_bf16_wmma<EPI_ADD><<<dim3(H / 128, M / 128), 256, 0, stream>>>(
      rwkv, WoT, h1, nullptr, h0, nullptr, M, H, A);

  // 9) ln2 -> shift -> mixes (kx2/rx2 bf16)
  ln_shift_mix<false><<<M, 256, 0, stream>>>(h1, ln2_w, ln2_b, nullptr, nullptr,
                                             f_tmk, nullptr, f_tmr, nullptr,
                                             kx2, nullptr, rx2, T, H);

  // 10) r2 = sigmoid(rx2 @ f_Wr)
  gemm_bf16_wmma<EPI_SIGMOID><<<dim3(H / 128, M / 128), 256, 0, stream>>>(
      rx2, fWrT, r2, nullptr, nullptr, nullptr, M, H, H);

  // 11) kk2 = relu(kx2 @ f_Wk)^2  (bf16)
  gemm_bf16_wmma<EPI_RELUSQ_BF16><<<dim3(F / 128, M / 128), 256, 0, stream>>>(
      kx2, fWkT, nullptr, kk2, nullptr, nullptr, M, F, H);

  // 12) out = h1 + r2 * (kk2 @ f_Wv)
  gemm_bf16_wmma<EPI_ADD_MUL><<<dim3(H / 128, M / 128), 256, 0, stream>>>(
      kk2, fWvT, out, nullptr, h1, r2, M, H, F);

  (void)in_sizes; (void)n_in; (void)out_size; (void)ws_size;
}
